// DemoLSTM_22634477650511
// MI455X (gfx1250) — matrix operand order repaired and checked by reading, not compiled, not run on hardware
//
#include <hip/hip_runtime.h>

typedef __bf16 bf16;
typedef __attribute__((ext_vector_type(16))) __bf16 v16bf;
typedef __attribute__((ext_vector_type(8)))  __bf16 v8bf;
typedef __attribute__((ext_vector_type(8)))  float  v8f;

#define Hh   51      // hidden size
#define G4   204     // 4*H gates
#define NP   208     // padded gate width (13 tiles of 16)
#define NT   13      // N tiles
#define T0   512     // sequence length
#define FUT  32      // future steps
#define TF   (T0 + FUT)
#define AS2  136     // A2 row stride (bf16): 68 dwords -> conflict-free b128 loads
#define BT   16      // batch rows per workgroup

__device__ __forceinline__ float fast_exp2(float x) {
#if __has_builtin(__builtin_amdgcn_exp2f)
  return __builtin_amdgcn_exp2f(x);
#else
  return exp2f(x);
#endif
}
__device__ __forceinline__ float sigmoid_(float x) {
  return 1.0f / (1.0f + fast_exp2(x * -1.4426950408889634f));
}
__device__ __forceinline__ float tanh_(float x) {
#if __has_builtin(__builtin_amdgcn_tanhf)
  return __builtin_amdgcn_tanhf(x);
#elif __has_builtin(__builtin_amdgcn_tanh_f32)
  return __builtin_amdgcn_tanh_f32(x);
#else
  return 1.0f - 2.0f / (1.0f + fast_exp2(x * 2.8853900817779268f));
#endif
}

// 16x32 bf16 A-fragment for v_wmma_f32_16x16x32_bf16 from a row-major LDS tile.
// lane (l&15)=M; elems 0..7 <-> K = 8*(l>>4)+0..7, elems 8..15 <-> K = 16+8*(l>>4)+0..7.
// p points at &A[M][ks*32 + 8*(l>>4)], 16B aligned.
__device__ __forceinline__ v16bf load_afrag(const bf16* p) {
  v8bf lo = *(const v8bf*)p;
  v8bf hi = *(const v8bf*)(p + 16);
  v16bf a;
#pragma unroll
  for (int e = 0; e < 8; ++e) { a[e] = lo[e]; a[e + 8] = hi[e]; }
  return a;
}

__global__ __launch_bounds__(256)
void lstm_seq_kernel(const float* __restrict__ x,
                     const float* __restrict__ Wih1, const float* __restrict__ Whh1,
                     const float* __restrict__ bih1, const float* __restrict__ bhh1,
                     const float* __restrict__ Wih2, const float* __restrict__ Whh2,
                     const float* __restrict__ bih2, const float* __restrict__ bhh2,
                     const float* __restrict__ Wlin, const float* __restrict__ blin,
                     float* __restrict__ out)
{
  __shared__ __align__(16) float g1buf[BT][NP];   // layer1 gate partials (for step t+1)
  __shared__ __align__(16) float g2buf[BT][NP];   // layer2 gate partials (for step t)
  __shared__ __align__(16) bf16  A2[BT][AS2];     // [h1 | h2], K padded to 128
  __shared__ __align__(16) bf16  xs[BT][T0];      // staged inputs (bf16)
  __shared__ __align__(16) float outbuf[BT][16];  // output ring, flushed every 16 steps

  const int tid  = threadIdx.x;
  const int wave = tid >> 5;
  const int lane = tid & 31;
  const int hh   = (lane >> 4) & 1;   // half-wave id (WMMA operand layout)
  const int ln   = lane & 15;
  const int m    = tid >> 4;          // elementwise: batch row owned by this half-wave
  const int l16  = tid & 15;          // elementwise: lane within half-wave
  const int b0   = blockIdx.x * BT;

  // ---------------- init LDS ----------------
  for (int i = tid; i < BT * AS2; i += 256) (&A2[0][0])[i] = (bf16)0.0f;
  for (int i = tid; i < BT * T0; i += 256) {
    const int mm = i >> 9, tt = i & (T0 - 1);
    xs[mm][tt] = (bf16)x[(size_t)(b0 + mm) * T0 + tt];   // coalesced over tt
  }
  const float blin0 = blin[0];

  // ------- per-thread register constants for the elementwise phase -------
  float b1r[4][4], w1r[4][4], b2r[4][4], wlr[4];
  float c1r[4] = {0.f, 0.f, 0.f, 0.f};
  float c2r[4] = {0.f, 0.f, 0.f, 0.f};
#pragma unroll
  for (int k = 0; k < 4; ++k) {
    const int j = l16 + 16 * k;
    const bool v = (j < Hh);
    wlr[k] = v ? Wlin[j] : 0.f;
#pragma unroll
    for (int g = 0; g < 4; ++g) {
      const int n = j + g * Hh;
      b1r[k][g] = v ? (bih1[n] + bhh1[n]) : 0.f;
      w1r[k][g] = v ? Wih1[n] : 0.f;
      b2r[k][g] = v ? (bih2[n] + bhh2[n]) : 0.f;
    }
  }

  // ------- per-wave B fragments in registers (constant over all steps) -------
  // B1 rows K>=51 are ZERO, which lets layer-1 WMMA reuse A2 fragments
  // (A2 cols 51..63 hold finite h2 values that multiply zero weights).
  const int t0i = wave;
  const int t1i = wave + 8;
  const int ntile = (t1i < NT) ? 2 : 1;

  v16bf B1[2][2];   // layer1: Whh1^T  [K=64][N]
  v16bf B2[2][4];   // layer2: [Wih2^T ; Whh2^T]  [K=128][N]
#pragma unroll
  for (int tt = 0; tt < 2; ++tt) {
    int tile = (tt == 0) ? t0i : t1i;
    if (tile >= NT) tile = t0i;                 // dup for uniformity; unused
    const int N = tile * 16 + ln;
#pragma unroll
    for (int ks = 0; ks < 2; ++ks) {
      v16bf f;
#pragma unroll
      for (int e = 0; e < 16; ++e) {
        const int K = ks * 32 + hh * 8 + e + (e >= 8 ? 8 : 0);
        float v = (K < Hh && N < G4) ? Whh1[N * Hh + K] : 0.f;
        f[e] = (bf16)v;
      }
      B1[tt][ks] = f;
    }
#pragma unroll
    for (int ks = 0; ks < 4; ++ks) {
      v16bf f;
#pragma unroll
      for (int e = 0; e < 16; ++e) {
        const int K = ks * 32 + hh * 8 + e + (e >= 8 ? 8 : 0);
        float v = 0.f;
        if (N < G4) {
          if (K < Hh)          v = Wih2[N * Hh + K];
          else if (K < 2 * Hh) v = Whh2[N * Hh + (K - Hh)];
        }
        f[e] = (bf16)v;
      }
      B2[tt][ks] = f;
    }
  }
  __syncthreads();

  // ---- prologue: h1(0) from x(0) alone (h1(-1)=0 => gates1(0)=0) ----
  {
    const float xv = (float)xs[m][0];
#pragma unroll
    for (int k = 0; k < 4; ++k) {
      const int j = l16 + 16 * k;
      if (j < Hh) {
        const float vi = b1r[k][0] + xv * w1r[k][0];
        const float vf = b1r[k][1] + xv * w1r[k][1];
        const float vg = b1r[k][2] + xv * w1r[k][2];
        const float vo = b1r[k][3] + xv * w1r[k][3];
        const float c  = sigmoid_(vf) * c1r[k] + sigmoid_(vi) * tanh_(vg);
        c1r[k] = c;
        const float h = sigmoid_(vo) * tanh_(c);
        A2[m][j] = (bf16)h;
      }
    }
  }
  __syncthreads();

  // ---------------- recurrence: 2 barriers per step ----------------
  for (int t = 0; t < TF; ++t) {
    // ---- W(t): gates2(t) = [h1(t)|h2(t-1)]@WB2 ; gates1(t+1) = h1(t)@Whh1^T ----
    // Flush the output ring (written during EL(t-16..t-1); fenced by last barrier).
    if (t != 0 && (t & 15) == 0)
      out[(size_t)(b0 + m) * TF + (t - 16) + l16] = outbuf[m][l16];
    {
      v16bf a0 = load_afrag(&A2[ln][0 * 32 + hh * 8]);
      v16bf a1 = load_afrag(&A2[ln][1 * 32 + hh * 8]);
      v16bf a2 = load_afrag(&A2[ln][2 * 32 + hh * 8]);
      v16bf a3 = load_afrag(&A2[ln][3 * 32 + hh * 8]);
      for (int tt = 0; tt < ntile; ++tt) {
        const int tile = (tt == 0) ? t0i : t1i;
        v8f d2 = {};
        d2 = __builtin_amdgcn_wmma_f32_16x16x32_bf16(false, a0, false, B2[tt][0], (short)0, d2, false, false);
        d2 = __builtin_amdgcn_wmma_f32_16x16x32_bf16(false, a1, false, B2[tt][1], (short)0, d2, false, false);
        d2 = __builtin_amdgcn_wmma_f32_16x16x32_bf16(false, a2, false, B2[tt][2], (short)0, d2, false, false);
        d2 = __builtin_amdgcn_wmma_f32_16x16x32_bf16(false, a3, false, B2[tt][3], (short)0, d2, false, false);
        v8f d1 = {};
        d1 = __builtin_amdgcn_wmma_f32_16x16x32_bf16(false, a0, false, B1[tt][0], (short)0, d1, false, false);
        d1 = __builtin_amdgcn_wmma_f32_16x16x32_bf16(false, a1, false, B1[tt][1], (short)0, d1, false, false);
#pragma unroll
        for (int v = 0; v < 8; ++v) {
          g2buf[8 * hh + v][tile * 16 + ln] = d2[v];
          g1buf[8 * hh + v][tile * 16 + ln] = d1[v];
        }
      }
    }
    __syncthreads();

    // ---- EL(t): layer-2 step t (+ output head), then layer-1 step t+1 ----
    float p = 0.f;
#pragma unroll
    for (int k = 0; k < 4; ++k) {
      const int j = l16 + 16 * k;
      if (j < Hh) {
        const float vi = g2buf[m][j         ] + b2r[k][0];
        const float vf = g2buf[m][j +     Hh] + b2r[k][1];
        const float vg = g2buf[m][j + 2 * Hh] + b2r[k][2];
        const float vo = g2buf[m][j + 3 * Hh] + b2r[k][3];
        const float c  = sigmoid_(vf) * c2r[k] + sigmoid_(vi) * tanh_(vg);
        c2r[k] = c;
        const float h = sigmoid_(vo) * tanh_(c);
        A2[m][Hh + j] = (bf16)h;   // h2(t) for step t+1
        p += h * wlr[k];
      }
    }
    // xor-reduce leaves the full sum in EVERY lane of the half-wave
    p += __shfl_xor(p, 1);
    p += __shfl_xor(p, 2);
    p += __shfl_xor(p, 4);
    p += __shfl_xor(p, 8);
    const float s = p + blin0;               // out(t), available in-register to all lanes
    if (l16 == 0) outbuf[m][t & 15] = s;

    {  // layer-1 for step t+1 (gates1(t+1) already in g1buf from W(t))
      const float xv = (t + 1 < T0) ? (float)xs[m][t + 1] : s;
#pragma unroll
      for (int k = 0; k < 4; ++k) {
        const int j = l16 + 16 * k;
        if (j < Hh) {
          const float vi = g1buf[m][j         ] + b1r[k][0] + xv * w1r[k][0];
          const float vf = g1buf[m][j +     Hh] + b1r[k][1] + xv * w1r[k][1];
          const float vg = g1buf[m][j + 2 * Hh] + b1r[k][2] + xv * w1r[k][2];
          const float vo = g1buf[m][j + 3 * Hh] + b1r[k][3] + xv * w1r[k][3];
          const float c  = sigmoid_(vf) * c1r[k] + sigmoid_(vi) * tanh_(vg);
          c1r[k] = c;
          const float h = sigmoid_(vo) * tanh_(c);
          A2[m][j] = (bf16)h;      // h1(t+1)
        }
      }
    }
    __syncthreads();
  }

  // final flush of the last 16 outputs (TF is a multiple of 16)
  out[(size_t)(b0 + m) * TF + (TF - 16) + l16] = outbuf[m][l16];
}

extern "C" void kernel_launch(void* const* d_in, const int* in_sizes, int n_in,
                              void* d_out, int out_size, void* d_ws, size_t ws_size,
                              hipStream_t stream) {
  (void)in_sizes; (void)n_in; (void)out_size; (void)d_ws; (void)ws_size;
  const float* x    = (const float*)d_in[0];
  const float* Wih1 = (const float*)d_in[1];
  const float* Whh1 = (const float*)d_in[2];
  const float* bih1 = (const float*)d_in[3];
  const float* bhh1 = (const float*)d_in[4];
  const float* Wih2 = (const float*)d_in[5];
  const float* Whh2 = (const float*)d_in[6];
  const float* bih2 = (const float*)d_in[7];
  const float* bhh2 = (const float*)d_in[8];
  const float* Wlin = (const float*)d_in[9];
  const float* blin = (const float*)d_in[10];
  lstm_seq_kernel<<<1024 / BT, 256, 0, stream>>>(
      x, Wih1, Whh1, bih1, bhh1, Wih2, Whh2, bih2, bhh2, Wlin, blin,
      (float*)d_out);
}
